// Spatial_Self_Attention_3126736192188
// MI455X (gfx1250) — compile-verified, matrix operand order checked by reading, not run on hardware
//
#include <hip/hip_runtime.h>
#include <hip/hip_bf16.h>

// ---------------------------------------------------------------------------
// Problem constants (from reference): B=4, DIM=384, H=W=48 -> N=2304 tokens,
// NH=8 heads, DEPTH=48, QKV dim = 1152.
// ---------------------------------------------------------------------------
static constexpr int BATCH = 4;
static constexpr int DIMC  = 384;
static constexpr int HW    = 48 * 48;       // 2304 tokens per batch
static constexpr int NTOK  = BATCH * HW;    // 9216 tokens total
static constexpr int NH    = 8;
static constexpr int DEPTH = 48;            // head dim
static constexpr int DPAD  = 64;            // head dim padded for K=32 WMMA loop
static constexpr int QKVD  = 3 * DIMC;      // 1152
static constexpr float EPSV = 1e-5f;

typedef __attribute__((ext_vector_type(16))) __bf16 v16bf;
typedef __attribute__((ext_vector_type(8)))  float  v8f;

// ---------------------------------------------------------------------------
// helpers
// ---------------------------------------------------------------------------
static __device__ __forceinline__ unsigned short f2bf_u16(float f) {
  unsigned u = __builtin_bit_cast(unsigned, f);
  unsigned r = u + 0x7fffu + ((u >> 16) & 1u);  // round-to-nearest-even
  return (unsigned short)(r >> 16);
}

static __device__ __forceinline__ unsigned pack_bf2(float lo, float hi) {
  return (unsigned)f2bf_u16(lo) | ((unsigned)f2bf_u16(hi) << 16);
}

// A / B^T fragment loader: memory holds a (16 x K) row-major slab (row = M for
// the A operand, row = N for the B operand loaded from B^T), 16-bit elements.
// Layout per CDNA5 ISA 7.12.2 (16-bit A 16x32): lane = {half, M&15};
// vgpr v holds K pair kb = (v<4 ? 2v : 16+2(v-4)) + 8*half.
// Per lane this is two contiguous 16B runs -> compiler emits ds_load_b128 x2.
static __device__ __forceinline__ v16bf load_frag_nk(const unsigned short* base, int ld) {
  const int lane = (int)(threadIdx.x & 31u);
  const int r  = lane & 15;
  const int rh = lane >> 4;
  v16bf f;
#pragma unroll
  for (int v = 0; v < 8; ++v) {
    const int kb = ((v < 4) ? (2 * v) : (16 + 2 * (v - 4))) + 8 * rh;
    f[2 * v]     = __builtin_bit_cast(__bf16, base[r * ld + kb]);
    f[2 * v + 1] = __builtin_bit_cast(__bf16, base[r * ld + kb + 1]);
  }
  return f;
}

// B fragment loader when memory holds B itself (K x N) row-major:
// element (k, n) at base[k*ld + n].
static __device__ __forceinline__ v16bf load_frag_kn(const unsigned short* base, int ld) {
  const int lane = (int)(threadIdx.x & 31u);
  const int n  = lane & 15;
  const int rh = lane >> 4;
  v16bf f;
#pragma unroll
  for (int v = 0; v < 8; ++v) {
    const int kb = ((v < 4) ? (2 * v) : (16 + 2 * (v - 4))) + 8 * rh;
    f[2 * v]     = __builtin_bit_cast(__bf16, base[kb * ld + n]);
    f[2 * v + 1] = __builtin_bit_cast(__bf16, base[(kb + 1) * ld + n]);
  }
  return f;
}

static __device__ __forceinline__ v8f wmma_bf16(v16bf a, v16bf b, v8f c) {
  // D = A(16x32) * B(32x16) + C, f32 accumulate -> v_wmma_f32_16x16x32_bf16
  return __builtin_amdgcn_wmma_f32_16x16x32_bf16(false, a, false, b, (short)0, c,
                                                 false, false);
}

// ---------------------------------------------------------------------------
// Kernel 1: per-token LayerNorm over channels; [B,C,H,W] -> bf16 x_ln[tok][C]
// ---------------------------------------------------------------------------
__global__ void ln_kernel(const float* __restrict__ feat,
                          const float* __restrict__ gamma,
                          const float* __restrict__ beta,
                          unsigned short* __restrict__ xln) {
  const int tok = (int)(blockIdx.x * blockDim.x + threadIdx.x);
  if (tok >= NTOK) return;
  const int b = tok / HW;
  const int n = tok - b * HW;
  const float* fb = feat + (size_t)b * DIMC * HW + n;   // stride HW per channel
  float s = 0.f, ss = 0.f;
  for (int c = 0; c < DIMC; ++c) {
    const float v = fb[(size_t)c * HW];
    s += v; ss += v * v;
  }
  const float mean = s * (1.0f / DIMC);
  const float var  = ss * (1.0f / DIMC) - mean * mean;
  const float inv  = rsqrtf(var + EPSV);
  unsigned short* xr = xln + (size_t)tok * DIMC;
  for (int c = 0; c < DIMC; c += 2) {
    const float v0 = (fb[(size_t)c * HW] - mean) * inv * gamma[c] + beta[c];
    const float v1 = (fb[(size_t)(c + 1) * HW] - mean) * inv * gamma[c + 1] + beta[c + 1];
    *reinterpret_cast<unsigned*>(xr + c) = pack_bf2(v0, v1);  // 4B-aligned
  }
}

// ---------------------------------------------------------------------------
// Kernel 2: convert weights f32 -> bf16 (packed 2 per thread)
// ---------------------------------------------------------------------------
__global__ void cvt_w_kernel(const float* __restrict__ wqkv,
                             const float* __restrict__ wout,
                             unsigned short* __restrict__ wqkv_bf,
                             unsigned short* __restrict__ wout_bf) {
  const int i2 = 2 * (int)(blockIdx.x * blockDim.x + threadIdx.x);
  const int NQ = QKVD * DIMC;        // 442368 (even)
  const int NO = DIMC * DIMC;        // 147456 (even)
  if (i2 < NQ) {
    *reinterpret_cast<unsigned*>(wqkv_bf + i2) = pack_bf2(wqkv[i2], wqkv[i2 + 1]);
  } else if (i2 < NQ + NO) {
    const int j = i2 - NQ;
    *reinterpret_cast<unsigned*>(wout_bf + j) = pack_bf2(wout[j], wout[j + 1]);
  }
}

// ---------------------------------------------------------------------------
// Kernels 3/5 shared GEMM body: C[M,ncols] = A[M,384](bf16) @ W^T, W:[ncols,384]
// Block: 128(M) x 64(N) tile, 256 threads = 8 waves (4x2), 32x32 per wave.
// Double-buffered LDS, uint4 (b128) staging, one barrier per K-step.
// Template epilogue flag: 0 -> bf16 store row-major [M,ldc];
//                         1 -> f32 + bias, transposed store to [B,C,H,W].
// ---------------------------------------------------------------------------
template <int EPILOGUE>
__device__ __forceinline__ void gemm_body(const unsigned short* __restrict__ A,
                                          const unsigned short* __restrict__ W,
                                          unsigned short* __restrict__ Cbf, int ldc,
                                          const float* __restrict__ bias,
                                          float* __restrict__ Cout) {
  __shared__ alignas(16) unsigned short As[2][128 * 32];
  __shared__ alignas(16) unsigned short Bs[2][64 * 32];
  const int m0 = (int)blockIdx.x * 128;
  const int n0 = (int)blockIdx.y * 64;
  const int tid = (int)threadIdx.x;
  const int wave = tid >> 5;
  const int waveM = wave >> 1, waveN = wave & 1;

  const uint4* A4 = reinterpret_cast<const uint4*>(A);
  const uint4* W4 = reinterpret_cast<const uint4*>(W);
  uint4 ra0, ra1, rb;

  // per-thread fetch of tile kt into registers (A: 512 uint4, B: 256 uint4)
  auto fetch = [&](int kt) {
    {
      const int r = tid >> 2, sl = tid & 3;
      ra0 = A4[(size_t)(m0 + r) * 48 + kt * 4 + sl];
      ra1 = A4[(size_t)(m0 + 64 + r) * 48 + kt * 4 + sl];
      rb  = W4[(size_t)(n0 + r) * 48 + kt * 4 + sl];
    }
  };
  auto stage = [&](int p) {
    uint4* As4 = reinterpret_cast<uint4*>(As[p]);
    uint4* Bs4 = reinterpret_cast<uint4*>(Bs[p]);
    As4[tid]       = ra0;
    As4[tid + 256] = ra1;
    Bs4[tid]       = rb;
  };

  const v8f vzero = {};
  v8f acc[2][2];
#pragma unroll
  for (int mi = 0; mi < 2; ++mi)
#pragma unroll
    for (int ni = 0; ni < 2; ++ni) acc[mi][ni] = vzero;

  fetch(0);
  stage(0);
  __syncthreads();

  for (int kt = 0; kt < DIMC / 32; ++kt) {
    const int p = kt & 1;
    if (kt + 1 < DIMC / 32) fetch(kt + 1);   // overlap HBM/L2 with WMMA
#pragma unroll
    for (int mi = 0; mi < 2; ++mi) {
      const v16bf a = load_frag_nk(&As[p][(waveM * 32 + mi * 16) * 32], 32);
#pragma unroll
      for (int ni = 0; ni < 2; ++ni) {
        const v16bf bb = load_frag_nk(&Bs[p][(waveN * 32 + ni * 16) * 32], 32);
        acc[mi][ni] = wmma_bf16(a, bb, acc[mi][ni]);
      }
    }
    if (kt + 1 < DIMC / 32) stage(1 - p);
    __syncthreads();
  }

  const int lane = tid & 31, cn = lane & 15, rh = lane >> 4;
#pragma unroll
  for (int mi = 0; mi < 2; ++mi)
#pragma unroll
    for (int ni = 0; ni < 2; ++ni) {
      const int row0 = m0 + waveM * 32 + mi * 16 + 8 * rh;  // 8 consecutive rows
      const int col  = n0 + waveN * 32 + ni * 16 + cn;
      if (EPILOGUE == 0) {
#pragma unroll
        for (int v = 0; v < 8; ++v)
          Cbf[(size_t)(row0 + v) * ldc + col] = f2bf_u16(acc[mi][ni][v]);
      } else {
        // rows are consecutive tokens of one batch; output layout [B,C,H,W]
        // puts them contiguous at fixed channel -> two float4 (b128) stores.
        const float bq = bias[col];
        const int bb = row0 / HW;
        const int n  = row0 - bb * HW;      // multiple of 8 -> 32B aligned
        float* dst = Cout + ((size_t)bb * DIMC + col) * HW + n;
        float4 f0 = make_float4(acc[mi][ni][0] + bq, acc[mi][ni][1] + bq,
                                acc[mi][ni][2] + bq, acc[mi][ni][3] + bq);
        float4 f1 = make_float4(acc[mi][ni][4] + bq, acc[mi][ni][5] + bq,
                                acc[mi][ni][6] + bq, acc[mi][ni][7] + bq);
        reinterpret_cast<float4*>(dst)[0] = f0;
        reinterpret_cast<float4*>(dst)[1] = f1;
      }
    }
}

__global__ void qkv_gemm_kernel(const unsigned short* __restrict__ xln,
                                const unsigned short* __restrict__ wq,
                                unsigned short* __restrict__ qkv) {
  gemm_body<0>(xln, wq, qkv, QKVD, nullptr, nullptr);
}

__global__ void out_gemm_kernel(const unsigned short* __restrict__ attn,
                                const unsigned short* __restrict__ wo,
                                const float* __restrict__ bout,
                                float* __restrict__ out) {
  gemm_body<1>(attn, wo, nullptr, 0, bout, out);
}

// ---------------------------------------------------------------------------
// Kernel 4: flash attention. grid = (18 q-tiles of 128, 8 heads, 4 batches),
// 256 threads = 8 waves; wave w owns Q rows [w*16, w*16+16) of the tile and
// the full 64-wide KV strip, so softmax row stats are wave-local.
// ---------------------------------------------------------------------------
__global__ void attn_kernel(const unsigned short* __restrict__ qkv,
                            const float* __restrict__ tau,
                            unsigned short* __restrict__ attn_out) {
  __shared__ alignas(16) unsigned short Qs[128 * DPAD];
  __shared__ alignas(16) unsigned short Ks[64 * DPAD];
  __shared__ alignas(16) unsigned short Vs[64 * DPAD];
  __shared__ alignas(16) unsigned short Ps[128 * 64];

  const int q0 = (int)blockIdx.x * 128;
  const int h  = (int)blockIdx.y;
  const int b  = (int)blockIdx.z;
  const int tid  = (int)threadIdx.x;
  const int wave = tid >> 5;
  const int lane = tid & 31, cn = lane & 15, rh = lane >> 4;
  const float scale = __expf(tau[h]);
  const size_t tokbase = (size_t)b * HW;

  // qkv row stride in uint4 units: 1152/8 = 144; head offset h*48/8 = h*6
  const uint4* qkv4 = reinterpret_cast<const uint4*>(qkv);
  const uint4 z4 = make_uint4(0u, 0u, 0u, 0u);

  // stage Q tile [128 x 48] -> [128 x 64] LDS via b128, zero-pad cols 48..63
  {
    uint4* Qs4 = reinterpret_cast<uint4*>(Qs);
    for (int i = tid; i < 128 * 8; i += 256) {       // 8 uint4 slots per row
      const int r = i >> 3, sl = i & 7;
      Qs4[i] = (sl < 6) ? qkv4[(tokbase + q0 + r) * 144 + h * 6 + sl] : z4;
    }
  }
  __syncthreads();

  const v16bf qa0 = load_frag_nk(&Qs[(wave * 16) * DPAD + 0],  DPAD);
  const v16bf qa1 = load_frag_nk(&Qs[(wave * 16) * DPAD + 32], DPAD);

  const v8f vzero = {};
  float m_i[8], l_i[8];
  v8f o[3];
#pragma unroll
  for (int v = 0; v < 8; ++v) { m_i[v] = -1e30f; l_i[v] = 0.f; }
#pragma unroll
  for (int t = 0; t < 3; ++t) o[t] = vzero;

  for (int kv = 0; kv < HW; kv += 64) {
    // stage K, V tiles [64 x 48] -> [64 x 64] LDS via b128
    {
      uint4* Ks4 = reinterpret_cast<uint4*>(Ks);
      uint4* Vs4 = reinterpret_cast<uint4*>(Vs);
      for (int i = tid; i < 64 * 8; i += 256) {
        const int r = i >> 3, sl = i & 7;
        if (sl < 6) {
          const size_t rb4 = (tokbase + kv + r) * 144 + h * 6 + sl;
          Ks4[i] = qkv4[rb4 + 48];   // +DIMC ushorts = +48 uint4
          Vs4[i] = qkv4[rb4 + 96];   // +2*DIMC ushorts = +96 uint4
        } else {
          Ks4[i] = z4; Vs4[i] = z4;
        }
      }
    }
    __syncthreads();

    // prefetch next KV tile while this one is consumed (global_prefetch_b8)
    if (kv + 64 < HW && tid < 128) {
      const int r = tid & 63;
      const size_t base = (tokbase + kv + 64 + r) * (size_t)QKVD + h * DEPTH;
      __builtin_prefetch(qkv + base + ((tid < 64) ? DIMC : 2 * DIMC), 0, 0);
    }

    // S = Q * K^T  (wave strip: 16 x 64, four 16x16 tiles, K-loop of 2)
    v8f s[4];
#pragma unroll
    for (int ct = 0; ct < 4; ++ct) {
      v8f accs = vzero;
      accs = wmma_bf16(qa0, load_frag_nk(&Ks[(ct * 16) * DPAD + 0],  DPAD), accs);
      accs = wmma_bf16(qa1, load_frag_nk(&Ks[(ct * 16) * DPAD + 32], DPAD), accs);
      s[ct] = accs;
    }

    // online softmax: accumulator row = v + 8*rh, cols spread over 16 lanes
#pragma unroll
    for (int v = 0; v < 8; ++v) {
      float mx = -1e30f;
#pragma unroll
      for (int ct = 0; ct < 4; ++ct) mx = fmaxf(mx, s[ct][v] * scale);
      mx = fmaxf(mx, __shfl_xor(mx, 1, 32));
      mx = fmaxf(mx, __shfl_xor(mx, 2, 32));
      mx = fmaxf(mx, __shfl_xor(mx, 4, 32));
      mx = fmaxf(mx, __shfl_xor(mx, 8, 32));
      const float mnew = fmaxf(m_i[v], mx);
      const float corr = __expf(m_i[v] - mnew);
      m_i[v] = mnew;
      float rsum = 0.f;
#pragma unroll
      for (int ct = 0; ct < 4; ++ct) {
        const float p = __expf(s[ct][v] * scale - mnew);
        s[ct][v] = p;
        rsum += p;
      }
      rsum += __shfl_xor(rsum, 1, 32);
      rsum += __shfl_xor(rsum, 2, 32);
      rsum += __shfl_xor(rsum, 4, 32);
      rsum += __shfl_xor(rsum, 8, 32);
      l_i[v] = l_i[v] * corr + rsum;
#pragma unroll
      for (int t = 0; t < 3; ++t) o[t][v] *= corr;
    }

    // round-trip P through LDS to re-fragment as an A operand
#pragma unroll
    for (int v = 0; v < 8; ++v) {
      const int pr = wave * 16 + v + 8 * rh;
#pragma unroll
      for (int ct = 0; ct < 4; ++ct)
        Ps[pr * 64 + ct * 16 + cn] = f2bf_u16(s[ct][v]);
    }
    __syncthreads();

    // O += P(16x64) * V(64x48)
#pragma unroll
    for (int kk = 0; kk < 2; ++kk) {
      const v16bf pa = load_frag_nk(&Ps[(wave * 16) * 64 + kk * 32], 64);
#pragma unroll
      for (int t = 0; t < 3; ++t) {
        const v16bf vb = load_frag_kn(&Vs[(kk * 32) * DPAD + t * 16], DPAD);
        o[t] = wmma_bf16(pa, vb, o[t]);
      }
    }
    __syncthreads();  // protect Ks/Vs/Ps before next tile
  }

  // normalize and write merged-head layout attn_out[tok][h*48 + d] (bf16)
#pragma unroll
  for (int v = 0; v < 8; ++v) {
    const float inv = 1.0f / l_i[v];
    const int row = q0 + wave * 16 + v + 8 * rh;
#pragma unroll
    for (int t = 0; t < 3; ++t) {
      const int col = h * DEPTH + t * 16 + cn;
      attn_out[(tokbase + row) * (size_t)DIMC + col] = f2bf_u16(o[t][v] * inv);
    }
  }
}

// ---------------------------------------------------------------------------
// launch
// ---------------------------------------------------------------------------
extern "C" void kernel_launch(void* const* d_in, const int* in_sizes, int n_in,
                              void* d_out, int out_size, void* d_ws, size_t ws_size,
                              hipStream_t stream) {
  (void)in_sizes; (void)n_in; (void)out_size; (void)ws_size;
  const float* feat  = (const float*)d_in[0];
  const float* gamma = (const float*)d_in[1];
  const float* beta  = (const float*)d_in[2];
  const float* tau   = (const float*)d_in[3];
  const float* wqkv  = (const float*)d_in[4];
  const float* wout  = (const float*)d_in[5];
  const float* bout  = (const float*)d_in[6];
  float* out = (float*)d_out;

  // workspace layout (bytes)
  char* ws = (char*)d_ws;
  const size_t off_xln  = 0;                                        // 9216*384*2
  const size_t off_wq   = off_xln  + (size_t)NTOK * DIMC * 2;       // 1152*384*2
  const size_t off_wo   = off_wq   + (size_t)QKVD * DIMC * 2;       // 384*384*2
  const size_t off_qkv  = off_wo   + (size_t)DIMC * DIMC * 2;       // 9216*1152*2
  const size_t off_attn = off_qkv  + (size_t)NTOK * QKVD * 2;       // 9216*384*2
  unsigned short* xln   = (unsigned short*)(ws + off_xln);
  unsigned short* wq_bf = (unsigned short*)(ws + off_wq);
  unsigned short* wo_bf = (unsigned short*)(ws + off_wo);
  unsigned short* qkvb  = (unsigned short*)(ws + off_qkv);
  unsigned short* attnb = (unsigned short*)(ws + off_attn);

  ln_kernel<<<NTOK / 256, 256, 0, stream>>>(feat, gamma, beta, xln);

  const int cvt_pairs = (QKVD * DIMC + DIMC * DIMC) / 2;
  cvt_w_kernel<<<(cvt_pairs + 255) / 256, 256, 0, stream>>>(wqkv, wout, wq_bf, wo_bf);

  qkv_gemm_kernel<<<dim3(NTOK / 128, QKVD / 64), 256, 0, stream>>>(xln, wq_bf, qkvb);

  attn_kernel<<<dim3(HW / 128, NH, BATCH), 256, 0, stream>>>(qkvb, tau, attnb);

  out_gemm_kernel<<<dim3(NTOK / 128, DIMC / 64), 256, 0, stream>>>(attnb, wo_bf, bout, out);
}
